// MultiHeadCausalAttention_91302414778937
// MI455X (gfx1250) — matrix operand order repaired and checked by reading, not compiled, not run on hardware
//
#include <hip/hip_runtime.h>
#include <hip/hip_bf16.h>

// ---------------------------------------------------------------------------
// CDNA5 (gfx1250, wave32) multi-head causal attention via v_wmma_f32_16x16x32_bf16
// ---------------------------------------------------------------------------

typedef __attribute__((ext_vector_type(16))) __bf16 v16bf;
typedef __attribute__((ext_vector_type(8)))  float  v8f;

union ABf {
  uint4 u[2];   // 32 bytes
  v16bf v;      // 16 x bf16
};

struct MaskOn  { static constexpr bool value = true;  };
struct MaskOff { static constexpr bool value = false; };

__device__ __forceinline__ unsigned short f2bf(float f) {
  unsigned int u = __float_as_uint(f);
  u += 0x7fffu + ((u >> 16) & 1u);   // round-to-nearest-even
  return (unsigned short)(u >> 16);
}

// ---------------------------------------------------------------------------
// Conversion kernels
// ---------------------------------------------------------------------------

__global__ void convert_f32_bf16(const float* __restrict__ in,
                                 unsigned short* __restrict__ out, int n) {
  int i = blockIdx.x * blockDim.x + threadIdx.x;
  if (i < n) out[i] = f2bf(in[i]);
}

// WT[n][k] = W[k][n] * scale  (W is [d_in, d_out] row-major; WT is [d_out][d_in])
__global__ void transpose_w_bf16(const float* __restrict__ W,
                                 unsigned short* __restrict__ WT,
                                 int Dd, float scale) {
  int i = blockIdx.x * blockDim.x + threadIdx.x;
  if (i >= Dd * Dd) return;
  int n = i / Dd;
  int k = i - n * Dd;
  WT[i] = f2bf(W[(size_t)k * Dd + n] * scale);
}

// ---------------------------------------------------------------------------
// bf16 GEMM: C[M,N] = A[M,K] * B[K,N], A row-major, B given transposed (N x K).
// 256 threads = 8 waves; block tile 128x128; each wave 32x64 (2x4 WMMA tiles).
// Epilogue MODE:
//   1: fp32 + bias, [M,N]
//   2: bf16, head-split  [B*H, S, 64]   (Q/K layout; hd = 64)
//   3: bf16, head-split + transposed [B*H, 64, S]  (V layout)
// ---------------------------------------------------------------------------

template <int MODE>
__global__ __launch_bounds__(256)
void gemm_bf16_wmma(const unsigned short* __restrict__ A,
                    const unsigned short* __restrict__ BT,
                    void* __restrict__ Cout,
                    const float* __restrict__ bias,
                    int M, int N, int Kd, int Sdim, int H) {
  const int lane = threadIdx.x & 31;
  const int wave = threadIdx.x >> 5;
  const int half = lane >> 4;
  const int l16  = lane & 15;
  const int m0 = blockIdx.y * 128 + (wave & 3) * 32;
  const int n0 = blockIdx.x * 128 + (wave >> 2) * 64;

  v8f acc[2][4];
#pragma unroll
  for (int i = 0; i < 2; ++i)
#pragma unroll
    for (int j = 0; j < 4; ++j) { v8f z = {}; acc[i][j] = z; }

  for (int kk = 0; kk < Kd; kk += 32) {
    ABf a[2], b[4];
    const int ca = kk + half * 8;          // A: VGPR0-3 = K ca..ca+7, VGPR4-7 = +16
#pragma unroll
    for (int i = 0; i < 2; ++i) {
      const unsigned short* p = A + (size_t)(m0 + i * 16 + l16) * Kd + ca;
      a[i].u[0] = *(const uint4*)(p);
      a[i].u[1] = *(const uint4*)(p + 16);
    }
    const int cb = kk + half * 8;          // B: VGPR0-3 = K cb..cb+7, VGPR4-7 = +16
#pragma unroll
    for (int j = 0; j < 4; ++j) {
      const unsigned short* p = BT + (size_t)(n0 + j * 16 + l16) * Kd + cb;
      b[j].u[0] = *(const uint4*)(p);
      b[j].u[1] = *(const uint4*)(p + 16);
    }
#pragma unroll
    for (int i = 0; i < 2; ++i)
#pragma unroll
      for (int j = 0; j < 4; ++j)
        acc[i][j] = __builtin_amdgcn_wmma_f32_16x16x32_bf16(
            false, a[i].v, false, b[j].v, (short)0, acc[i][j], false, false);
  }

#pragma unroll
  for (int i = 0; i < 2; ++i)
#pragma unroll
    for (int j = 0; j < 4; ++j)
#pragma unroll
      for (int r = 0; r < 8; ++r) {
        int row = m0 + i * 16 + r + 8 * half;   // C layout: M = r + 8*(lane/16)
        int col = n0 + j * 16 + l16;            //           N = lane%16
        float v = acc[i][j][r];
        if (MODE == 1) {
          ((float*)Cout)[(size_t)row * N + col] = v + bias[col];
        } else {
          int bidx = row / Sdim, s = row - bidx * Sdim;
          int h = col >> 6, d = col & 63;
          size_t bh = (size_t)bidx * H + h;
          if (MODE == 2)
            ((unsigned short*)Cout)[(bh * Sdim + s) * 64 + d] = f2bf(v);
          else
            ((unsigned short*)Cout)[(bh * 64 + d) * (size_t)Sdim + s] = f2bf(v);
        }
      }
}

// ---------------------------------------------------------------------------
// Flash attention (causal), hd = 64. One wave handles 16 query rows of one
// (b,h). Q pre-scaled by hd^-0.5 (folded into Wq). KV blocks of 64 keys.
//  - Steady-state blocks take a mask-free fast path (only diagonal block pays
//    for compares/cndmask).
//  - Row-sum of P is computed on the matrix pipe (P x ones-tile WMMA) instead
//    of an LDS-pipe shuffle butterfly; only the row-max uses ds_bpermute.
// Q,K: [B*H, S, 64] bf16 ; VT: [B*H, 64, S] bf16 ; O: [B*S, D] bf16.
// ---------------------------------------------------------------------------

__global__ __launch_bounds__(128)
void flash_attn_wmma(const unsigned short* __restrict__ Q,
                     const unsigned short* __restrict__ Kmat,
                     const unsigned short* __restrict__ VT,
                     unsigned short* __restrict__ O,
                     int S, int H, int D) {
  __shared__ __align__(16) unsigned short plds[4][16 * 64];  // 2 KB per wave

  const int lane = threadIdx.x & 31;
  const int wv   = threadIdx.x >> 5;
  const int half = lane >> 4;
  const int l16  = lane & 15;

  const int gw      = blockIdx.x * 4 + wv;
  const int rbPerBH = S / 16;
  const int bh = gw / rbPerBH;
  const int q0 = (gw - bh * rbPerBH) * 16;
  const int b = bh / H, h = bh - b * H;

  const unsigned short* Qb = Q    + (size_t)bh * S * 64;
  const unsigned short* Kb = Kmat + (size_t)bh * S * 64;
  const unsigned short* Vb = VT   + (size_t)bh * 64 * S;

  // Q as two 16x32 A-tiles (dims 0..31, 32..63)
  ABf qa[2];
#pragma unroll
  for (int ks = 0; ks < 2; ++ks) {
    const unsigned short* p = Qb + (size_t)(q0 + l16) * 64 + ks * 32 + half * 8;
    qa[ks].u[0] = *(const uint4*)(p);
    qa[ks].u[1] = *(const uint4*)(p + 16);
  }

  // all-ones 32x16 B-tile (bf16 1.0 = 0x3F80) for row-sum WMMA
  ABf ones;
  ones.u[0] = make_uint4(0x3F803F80u, 0x3F803F80u, 0x3F803F80u, 0x3F803F80u);
  ones.u[1] = ones.u[0];

  float mi[8], li[8];
  v8f accO[4];
#pragma unroll
  for (int r = 0; r < 8; ++r) { mi[r] = -1e30f; li[r] = 0.f; }
#pragma unroll
  for (int t = 0; t < 4; ++t) { v8f z = {}; accO[t] = z; }

  unsigned short* pl = plds[wv];

  auto process = [&](int kv0, auto mask_tag) {
    constexpr bool MASK = decltype(mask_tag)::value;

    // scores: S[16 x 64] = Q(16x64) * K^T ; four 16x16 N-tiles, K-depth 64
    v8f s[4];
#pragma unroll
    for (int nt = 0; nt < 4; ++nt) {
      v8f c = {};
#pragma unroll
      for (int ks = 0; ks < 2; ++ks) {
        ABf bk;
        const unsigned short* p =
            Kb + (size_t)(kv0 + nt * 16 + l16) * 64 + ks * 32 + half * 8;
        bk.u[0] = *(const uint4*)(p);
        bk.u[1] = *(const uint4*)(p + 16);
        c = __builtin_amdgcn_wmma_f32_16x16x32_bf16(
            false, qa[ks].v, false, bk.v, (short)0, c, false, false);
      }
      s[nt] = c;
    }

    if (MASK) {
#pragma unroll
      for (int nt = 0; nt < 4; ++nt)
#pragma unroll
        for (int r = 0; r < 8; ++r) {
          int key = kv0 + nt * 16 + l16;
          int qry = q0 + r + 8 * half;
          if (key > qry) s[nt][r] = -1e30f;
        }
    }

    // online softmax: row max via 4-level butterfly (16-lane halves)
    float alpha[8];
#pragma unroll
    for (int r = 0; r < 8; ++r) {
      float v = fmaxf(fmaxf(s[0][r], s[1][r]), fmaxf(s[2][r], s[3][r]));
#pragma unroll
      for (int mk = 8; mk >= 1; mk >>= 1) v = fmaxf(v, __shfl_xor(v, mk, 32));
      float mn = fmaxf(mi[r], v);
      alpha[r] = __expf(mi[r] - mn);
      mi[r] = mn;
#pragma unroll
      for (int nt = 0; nt < 4; ++nt) s[nt][r] = __expf(s[nt][r] - mn);
    }
#pragma unroll
    for (int t = 0; t < 4; ++t)
#pragma unroll
      for (int r = 0; r < 8; ++r) accO[t][r] *= alpha[r];

    // C-layout P -> row-major 16x64 bf16 in LDS -> reload in A-layout
#pragma unroll
    for (int nt = 0; nt < 4; ++nt)
#pragma unroll
      for (int r = 0; r < 8; ++r)
        pl[(r + 8 * half) * 64 + nt * 16 + l16] = f2bf(s[nt][r]);
    asm volatile("s_wait_dscnt 0" ::: "memory");  // per-wave; waves diverge, no barrier

    ABf pa[2];
#pragma unroll
    for (int ks = 0; ks < 2; ++ks) {
      const unsigned short* pp = pl + l16 * 64 + ks * 32 + half * 8;
      pa[ks].u[0] = *(const uint4*)(pp);
      pa[ks].u[1] = *(const uint4*)(pp + 16);
    }

    // row-sum of P on the matrix pipe: rs[r] = sum_k P[row][k]
    v8f rs = {};
    rs = __builtin_amdgcn_wmma_f32_16x16x32_bf16(
        false, pa[0].v, false, ones.v, (short)0, rs, false, false);
    rs = __builtin_amdgcn_wmma_f32_16x16x32_bf16(
        false, pa[1].v, false, ones.v, (short)0, rs, false, false);
#pragma unroll
    for (int r = 0; r < 8; ++r) li[r] = alpha[r] * li[r] + rs[r];

    // O += P(16x64) * V(64x64): four 16x16 N-tiles, K-depth 64, VT rows = dims
#pragma unroll
    for (int ks = 0; ks < 2; ++ks)
#pragma unroll
      for (int t = 0; t < 4; ++t) {
        ABf bv;
        const unsigned short* p =
            Vb + (size_t)(t * 16 + l16) * S + kv0 + ks * 32 + half * 8;
        bv.u[0] = *(const uint4*)(p);
        bv.u[1] = *(const uint4*)(p + 16);
        accO[t] = __builtin_amdgcn_wmma_f32_16x16x32_bf16(
            false, pa[ks].v, false, bv.v, (short)0, accO[t], false, false);
      }
  };

  const int nbFull = q0 / 64;        // blocks with kv0+63 <= q0: mask-free
  const int nb     = (q0 + 79) / 64; // covers keys 0 .. q0+15
  for (int j = 0; j < nbFull; ++j) process(j * 64, MaskOff{});
  for (int j = nbFull; j < nb; ++j) process(j * 64, MaskOn{});

  // normalize + store into [B*S, D] layout (head h occupies cols h*64..h*64+63)
#pragma unroll
  for (int r = 0; r < 8; ++r) {
    float inv = 1.0f / li[r];
    int srow = q0 + r + 8 * half;
    size_t orow = (size_t)b * S + srow;
#pragma unroll
    for (int t = 0; t < 4; ++t)
      O[orow * D + h * 64 + t * 16 + l16] = f2bf(accO[t][r] * inv);
  }
}

// ---------------------------------------------------------------------------
// Launch
// ---------------------------------------------------------------------------

extern "C" void kernel_launch(void* const* d_in, const int* in_sizes, int n_in,
                              void* d_out, int out_size, void* d_ws, size_t ws_size,
                              hipStream_t stream) {
  const float* x  = (const float*)d_in[0];
  const float* Wq = (const float*)d_in[1];
  const float* Wk = (const float*)d_in[2];
  const float* Wv = (const float*)d_in[3];
  const float* Wo = (const float*)d_in[4];
  const float* bo = (const float*)d_in[5];
  float* out = (float*)d_out;

  const int B = 2, S = 2048, D = 1024, H = 16;
  const int Mrows = B * S;           // 4096
  const int ND = Mrows * D;          // 4 Mi elements

  char* ws = (char*)d_ws;
  unsigned short* xb  = (unsigned short*)(ws + (size_t)0);           // 8 MiB
  unsigned short* WqT = (unsigned short*)(ws + (size_t)(8u  << 20)); // 2 MiB each
  unsigned short* WkT = (unsigned short*)(ws + (size_t)(10u << 20));
  unsigned short* WvT = (unsigned short*)(ws + (size_t)(12u << 20));
  unsigned short* WoT = (unsigned short*)(ws + (size_t)(14u << 20));
  unsigned short* Qb  = (unsigned short*)(ws + (size_t)(16u << 20)); // 8 MiB each
  unsigned short* Kb  = (unsigned short*)(ws + (size_t)(24u << 20));
  unsigned short* Vt  = (unsigned short*)(ws + (size_t)(32u << 20));
  unsigned short* Ab  = (unsigned short*)(ws + (size_t)(40u << 20));

  const float qscale = 0.125f;       // hd^-0.5 folded into Wq

  convert_f32_bf16<<<ND / 256, 256, 0, stream>>>(x, xb, ND);
  transpose_w_bf16<<<(D * D) / 256, 256, 0, stream>>>(Wq, WqT, D, qscale);
  transpose_w_bf16<<<(D * D) / 256, 256, 0, stream>>>(Wk, WkT, D, 1.0f);
  transpose_w_bf16<<<(D * D) / 256, 256, 0, stream>>>(Wv, WvT, D, 1.0f);
  transpose_w_bf16<<<(D * D) / 256, 256, 0, stream>>>(Wo, WoT, D, 1.0f);

  dim3 ggrid(D / 128, Mrows / 128);  // (8, 32)

  // projections with fused head-split epilogues
  gemm_bf16_wmma<2><<<ggrid, 256, 0, stream>>>(xb, WqT, Qb, nullptr, Mrows, D, D, S, H);
  gemm_bf16_wmma<2><<<ggrid, 256, 0, stream>>>(xb, WkT, Kb, nullptr, Mrows, D, D, S, H);
  gemm_bf16_wmma<3><<<ggrid, 256, 0, stream>>>(xb, WvT, Vt, nullptr, Mrows, D, D, S, H);

  // causal flash attention: B*H*(S/16) waves, 4 waves / block
  const int nwaves = B * H * (S / 16);   // 4096
  flash_attn_wmma<<<nwaves / 4, 128, 0, stream>>>(Qb, Kb, Vt, Ab, S, H, D);

  // output projection + bias -> fp32
  gemm_bf16_wmma<1><<<ggrid, 256, 0, stream>>>(Ab, WoT, out, bo, Mrows, D, D, S, H);
}
